// CLIP_13967233646688
// MI455X (gfx1250) — matrix operand order repaired and checked by reading, not compiled, not run on hardware
//
#include <hip/hip_runtime.h>
#include <hip/hip_bf16.h>

// ---------------------------------------------------------------------------
// Types for CDNA5 WMMA (gfx1250, wave32)
// ---------------------------------------------------------------------------
typedef __bf16 bf16_t;
typedef __attribute__((ext_vector_type(8)))  __bf16 v8bf;
typedef __attribute__((ext_vector_type(16))) __bf16 v16bf;
typedef __attribute__((ext_vector_type(8)))  float  v8f;
typedef __attribute__((ext_vector_type(4)))  int    v4i;

#define NSEQ   512      // M*N sequences
#define LSEQ   77
#define DMODEL 768
#define RROWS  (NSEQ*LSEQ)   // 39424 rows, multiple of 64
#define NHEAD  12
#define HD     64

#ifndef __has_builtin
#define __has_builtin(x) 0
#endif
#if __has_builtin(__builtin_amdgcn_global_load_async_to_lds_b128)
#define ASYNC_COPY 1
#else
#define ASYNC_COPY 0
#endif

__device__ __forceinline__ bf16_t f2bf(float f) {
    unsigned u = __float_as_uint(f);
    unsigned r = (u + 0x7fffu + ((u >> 16) & 1u)) >> 16;
    return __builtin_bit_cast(bf16_t, (unsigned short)r);
}
__device__ __forceinline__ float bf2f(bf16_t b) {
    unsigned short s = __builtin_bit_cast(unsigned short, b);
    return __uint_as_float(((unsigned)s) << 16);
}

// 16-byte global -> LDS copy: async DMA path on CDNA5, sync fallback otherwise.
__device__ __forceinline__ void cp16_g2l(void* lds, const void* gl) {
#if ASYNC_COPY
    __builtin_amdgcn_global_load_async_to_lds_b128(
        (__attribute__((address_space(1))) v4i*)(gl),
        (__attribute__((address_space(3))) v4i*)(lds), 0, 0);
#else
    *(uint4*)lds = *(const uint4*)gl;
#endif
}
__device__ __forceinline__ void wait_async_5() {
#if ASYNC_COPY
#if __has_builtin(__builtin_amdgcn_s_wait_asynccnt)
    __builtin_amdgcn_s_wait_asynccnt(5);
#else
    asm volatile("s_wait_asynccnt 0x5" ::: "memory");
#endif
#endif
}
__device__ __forceinline__ void wait_async_0() {
#if ASYNC_COPY
#if __has_builtin(__builtin_amdgcn_s_wait_asynccnt)
    __builtin_amdgcn_s_wait_asynccnt(0);
#else
    asm volatile("s_wait_asynccnt 0x0" ::: "memory");
#endif
#endif
}

// ---------------------------------------------------------------------------
// fp32 -> bf16 bulk convert (weights)
// ---------------------------------------------------------------------------
__global__ __launch_bounds__(256) void cvt_bf16_kernel(const float* __restrict__ in,
                                                       bf16_t* __restrict__ out,
                                                       long long n) {
    long long i = (long long)blockIdx.x * 256 + threadIdx.x;
    long long stride = (long long)gridDim.x * 256;
    for (; i < n; i += stride) out[i] = f2bf(in[i]);
}

// ---------------------------------------------------------------------------
// find SOV / EOT positions per sequence (first occurrence == jnp.argmax of mask)
// ---------------------------------------------------------------------------
__global__ void find_pos_kernel(const int* __restrict__ text,
                                int* __restrict__ posy, int* __restrict__ eot) {
    int n = threadIdx.x;
    if (n >= 64) return;
    int py = -1, et = -1;
    for (int p = 0; p < LSEQ; ++p) {
        int t = text[n * LSEQ + p];
        if (py < 0 && t == 49408) py = p;
        if (et < 0 && t == 49407) et = p;
    }
    posy[n] = (py < 0) ? 0 : py;
    eot[n]  = (et < 0) ? 0 : et;
}

// ---------------------------------------------------------------------------
// Embedding: token gather with V-slot shift, visual feature splice, + pos_emb
// grid = (LSEQ, NSEQ)
// ---------------------------------------------------------------------------
__global__ __launch_bounds__(256) void embed_kernel(const int* __restrict__ text,
                                                    const float* __restrict__ visf,
                                                    const float* __restrict__ tok,
                                                    const float* __restrict__ pos,
                                                    const int* __restrict__ posy,
                                                    float* __restrict__ h) {
    int p = blockIdx.x;           // position 0..76
    int bidx = blockIdx.y;        // 0..511  (m*64 + n)
    int m = bidx >> 6, n = bidx & 63;
    int py = posy[n];
    bool isvis = (p >= py) && (p < py + 8);
    int src = (p < py) ? p : (p - 7);
    src = min(max(src, 0), LSEQ - 1);
    int vidx = min(max(p - py, 0), 7);
    int tokid = text[n * LSEQ + src];
    const float* vsrc = isvis ? (visf + ((size_t)m * 8 + vidx) * DMODEL)
                              : (tok + (size_t)tokid * DMODEL);
    float* hr = h + ((size_t)bidx * LSEQ + p) * DMODEL;
    const float* pr = pos + (size_t)p * DMODEL;
    for (int d = threadIdx.x; d < DMODEL; d += 256) hr[d] = vsrc[d] + pr[d];
}

// ---------------------------------------------------------------------------
// LayerNorm over D=768, fp32 in -> bf16 out (GEMM A operand).  1 block per row.
// ---------------------------------------------------------------------------
__global__ __launch_bounds__(256) void ln_bf16_kernel(const float* __restrict__ x,
                                                      const float* __restrict__ w,
                                                      const float* __restrict__ b,
                                                      bf16_t* __restrict__ y) {
    int r = blockIdx.x, tid = threadIdx.x;
    const float* xr = x + (size_t)r * DMODEL;
    float v0 = xr[tid], v1 = xr[tid + 256], v2 = xr[tid + 512];
    __shared__ float r1[256], r2[256];
    r1[tid] = v0 + v1 + v2;
    r2[tid] = v0 * v0 + v1 * v1 + v2 * v2;
    __syncthreads();
    for (int o = 128; o > 0; o >>= 1) {
        if (tid < o) { r1[tid] += r1[tid + o]; r2[tid] += r2[tid + o]; }
        __syncthreads();
    }
    float mu = r1[0] * (1.0f / DMODEL);
    float var = r2[0] * (1.0f / DMODEL) - mu * mu;
    float rs = rsqrtf(var + 1e-5f);
    bf16_t* yr = y + (size_t)r * DMODEL;
    yr[tid]       = f2bf((v0 - mu) * rs * w[tid]       + b[tid]);
    yr[tid + 256] = f2bf((v1 - mu) * rs * w[tid + 256] + b[tid + 256]);
    yr[tid + 512] = f2bf((v2 - mu) * rs * w[tid + 512] + b[tid + 512]);
}

// ---------------------------------------------------------------------------
// WMMA bf16 GEMM:  C[M x N] = A[M x K] * W[N x K]^T (+bias, epilogue)
//   MODE 0: bias -> bf16 out
//   MODE 1: bias + x*sigmoid(1.702x) -> bf16 out
//   MODE 2: bias + residual accumulate into fp32 out
// Block tile 64x256, 8 waves (2x4), wave tile 32x64, K-step 32.
// Double-buffered LDS staging via async global->LDS DMA (ASYNCcnt-tracked),
// overlapped with v_wmma compute.  grid = (N/256, M/64)
// ---------------------------------------------------------------------------
template <int MODE>
__global__ __launch_bounds__(256) void gemm_wmma_kernel(const bf16_t* __restrict__ A,
                                                        const bf16_t* __restrict__ W,
                                                        const float* __restrict__ bias,
                                                        void* __restrict__ outp,
                                                        int K, int Nout) {
    __shared__ __align__(16) bf16_t As[2][64 * 40];    // +8 bf16 pad per row
    __shared__ __align__(16) bf16_t Bs[2][256 * 40];
    const int tid = threadIdx.x;
    const int m0 = blockIdx.y * 64;
    const int n0 = blockIdx.x * 256;
    const int lane = tid & 31;
    const int wave = tid >> 5;
    const int wm = wave >> 2;          // 0..1
    const int wn = wave & 3;           // 0..3
    const int lrow = lane & 15;
    const int lhalf = lane >> 4;       // 0 | 1

    v8f acc[2][4] = {};

    const int ldrow = tid >> 2;        // 0..63
    const int ldc   = (tid & 3) * 8;   // 0,8,16,24

    // issue one tile's staging (5 x 16B per thread: 1 A + 4 B)
    auto issue = [&](int k0, int buf) {
        cp16_g2l(&As[buf][ldrow * 40 + ldc],
                 A + (size_t)(m0 + ldrow) * K + k0 + ldc);
#pragma unroll
        for (int i = 0; i < 4; ++i) {
            int br = ldrow + i * 64;
            cp16_g2l(&Bs[buf][br * 40 + ldc],
                     W + (size_t)(n0 + br) * K + k0 + ldc);
        }
    };

    const int nk = K >> 5;
    issue(0, 0);
    for (int kt = 0; kt < nk; ++kt) {
        const int buf = kt & 1;
        if (kt + 1 < nk) {
            issue((kt + 1) << 5, buf ^ 1);  // prefetch next tile into other buffer
            wait_async_5();                 // tile kt's 5 async ops retired (in-order)
        } else {
            wait_async_0();
        }
        __syncthreads();                    // all waves' tile-kt data visible in LDS

        // A fragment: lane holds row (lane&15); K chunks {b..b+7, b+16..b+23}, b = lhalf*8
        v16bf afr[2];
#pragma unroll
        for (int mi = 0; mi < 2; ++mi) {
            int r = wm * 32 + mi * 16 + lrow;
            v8bf lo = *(const v8bf*)&As[buf][r * 40 + lhalf * 8];
            v8bf hi = *(const v8bf*)&As[buf][r * 40 + lhalf * 8 + 16];
            afr[mi] = __builtin_shufflevector(lo, hi, 0,1,2,3,4,5,6,7,8,9,10,11,12,13,14,15);
        }
        // B fragment: lane holds column (lane&15); K chunks {b..b+7, b+16..b+23}, b = lhalf*8
        v16bf bfr[4];
#pragma unroll
        for (int ni = 0; ni < 4; ++ni) {
            int c = wn * 64 + ni * 16 + lrow;
            v8bf lo = *(const v8bf*)&Bs[buf][c * 40 + lhalf * 8];
            v8bf hi = *(const v8bf*)&Bs[buf][c * 40 + lhalf * 8 + 16];
            bfr[ni] = __builtin_shufflevector(lo, hi, 0,1,2,3,4,5,6,7,8,9,10,11,12,13,14,15);
        }
#pragma unroll
        for (int mi = 0; mi < 2; ++mi)
#pragma unroll
            for (int ni = 0; ni < 4; ++ni)
                acc[mi][ni] = __builtin_amdgcn_wmma_f32_16x16x32_bf16(
                    false, afr[mi], false, bfr[ni], (short)0, acc[mi][ni], false, false);
        __syncthreads();                    // WAR: done reading buf before it is re-filled
    }

    // Epilogue.  C/D layout: col = lane&15, row = e + 8*lhalf (per 16x16 tile).
#pragma unroll
    for (int mi = 0; mi < 2; ++mi) {
        int rbase = m0 + wm * 32 + mi * 16 + lhalf * 8;
#pragma unroll
        for (int ni = 0; ni < 4; ++ni) {
            int gc = n0 + wn * 64 + ni * 16 + lrow;
            float bv = bias[gc];
#pragma unroll
            for (int e = 0; e < 8; ++e) {
                float v = acc[mi][ni][e] + bv;
                size_t idx = (size_t)(rbase + e) * Nout + gc;
                if constexpr (MODE == 0) {
                    ((bf16_t*)outp)[idx] = f2bf(v);
                } else if constexpr (MODE == 1) {
                    float g = v / (1.0f + __expf(-1.702f * v));
                    ((bf16_t*)outp)[idx] = f2bf(g);
                } else {
                    ((float*)outp)[idx] += v;
                }
            }
        }
    }
}

// ---------------------------------------------------------------------------
// Attention: one block per (head, sequence).  fp32 math (~1% of FLOPs).
// qkv is bf16 [R, 2304] (q|k|v).  Output obf bf16 [R, 768] head-concatenated.
// grid = (NHEAD, NSEQ)
// ---------------------------------------------------------------------------
__global__ __launch_bounds__(256) void attention_kernel(const bf16_t* __restrict__ qkv,
                                                        bf16_t* __restrict__ obf) {
    int hh = blockIdx.x;       // head
    int b  = blockIdx.y;       // sequence
    __shared__ float kf[96 * 65];
    __shared__ float vf[96 * 65];
    int tid = threadIdx.x;
    const size_t base = (size_t)b * LSEQ * 2304;
    for (int idx = tid; idx < 96 * 64; idx += 256) {
        int row = idx >> 6, d = idx & 63;
        float kv = 0.f, vv = 0.f;
        if (row < LSEQ) {
            size_t o = base + (size_t)row * 2304 + hh * HD + d;
            kv = bf2f(qkv[o + 768]);
            vv = bf2f(qkv[o + 1536]);
        }
        kf[row * 65 + d] = kv;
        vf[row * 65 + d] = vv;
    }
    __syncthreads();

    int wave = tid >> 5, lane = tid & 31;
    for (int i = wave; i < LSEQ; i += 8) {
        size_t qo = base + (size_t)i * 2304 + hh * HD;
        float q0 = bf2f(qkv[qo + lane]);
        float q1 = bf2f(qkv[qo + 32 + lane]);
        int j0 = lane, j1 = lane + 32, j2 = lane + 64;
        float s0 = 0.f, s1 = 0.f, s2 = 0.f;
#pragma unroll 8
        for (int d = 0; d < 32; ++d) {
            float qd = __shfl(q0, d, 32);
            s0 += qd * kf[j0 * 65 + d];
            s1 += qd * kf[j1 * 65 + d];
            s2 += qd * kf[j2 * 65 + d];
        }
#pragma unroll 8
        for (int d = 0; d < 32; ++d) {
            float qd = __shfl(q1, d, 32);
            s0 += qd * kf[j0 * 65 + 32 + d];
            s1 += qd * kf[j1 * 65 + 32 + d];
            s2 += qd * kf[j2 * 65 + 32 + d];
        }
        const float NEG = -1e30f;
        s0 = (j0 <= i) ? s0 * 0.125f : NEG;
        s1 = (j1 <= i) ? s1 * 0.125f : NEG;
        s2 = (j2 <= i) ? s2 * 0.125f : NEG;
        float mx = fmaxf(s0, fmaxf(s1, s2));
        for (int o = 16; o > 0; o >>= 1) mx = fmaxf(mx, __shfl_xor(mx, o, 32));
        float p0 = (j0 <= i) ? __expf(s0 - mx) : 0.f;
        float p1 = (j1 <= i) ? __expf(s1 - mx) : 0.f;
        float p2 = (j2 <= i) ? __expf(s2 - mx) : 0.f;
        float sm = p0 + p1 + p2;
        for (int o = 16; o > 0; o >>= 1) sm += __shfl_xor(sm, o, 32);
        float inv = 1.0f / sm;
        p0 *= inv; p1 *= inv; p2 *= inv;
        float o0 = 0.f, o1 = 0.f;
        for (int j = 0; j <= i; ++j) {
            float pj;
            if (j < 32)      pj = __shfl(p0, j, 32);
            else if (j < 64) pj = __shfl(p1, j - 32, 32);
            else             pj = __shfl(p2, j - 64, 32);
            o0 += pj * vf[j * 65 + lane];
            o1 += pj * vf[j * 65 + 32 + lane];
        }
        size_t oo = ((size_t)b * LSEQ + i) * DMODEL + hh * HD;
        obf[oo + lane]      = f2bf(o0);
        obf[oo + 32 + lane] = f2bf(o1);
    }
}

// ---------------------------------------------------------------------------
// Final LN on gathered EOT rows -> fp32 featin[512,768].  grid = 512 blocks.
// ---------------------------------------------------------------------------
__global__ __launch_bounds__(256) void lnf_gather_kernel(const float* __restrict__ h,
                                                         const int* __restrict__ eot,
                                                         const float* __restrict__ w,
                                                         const float* __restrict__ b,
                                                         float* __restrict__ out) {
    int r = blockIdx.x, tid = threadIdx.x;   // r = m*64 + n
    int n = r & 63;
    int srow = r * LSEQ + (eot[n] + 7);      // eot + V - 1
    const float* xr = h + (size_t)srow * DMODEL;
    float v0 = xr[tid], v1 = xr[tid + 256], v2 = xr[tid + 512];
    __shared__ float r1[256], r2[256];
    r1[tid] = v0 + v1 + v2;
    r2[tid] = v0 * v0 + v1 * v1 + v2 * v2;
    __syncthreads();
    for (int o = 128; o > 0; o >>= 1) {
        if (tid < o) { r1[tid] += r1[tid + o]; r2[tid] += r2[tid + o]; }
        __syncthreads();
    }
    float mu = r1[0] * (1.0f / DMODEL);
    float var = r2[0] * (1.0f / DMODEL) - mu * mu;
    float rs = rsqrtf(var + 1e-5f);
    float* yr = out + (size_t)r * DMODEL;
    yr[tid]       = (v0 - mu) * rs * w[tid]       + b[tid];
    yr[tid + 256] = (v1 - mu) * rs * w[tid + 256] + b[tid + 256];
    yr[tid + 512] = (v2 - mu) * rs * w[tid + 512] + b[tid + 512];
}

// Small fp32 projection: feats[512,768] = featin @ text_proj[768,768]. grid=(3,512)
__global__ __launch_bounds__(256) void proj_kernel(const float* __restrict__ fin,
                                                   const float* __restrict__ tp,
                                                   float* __restrict__ feats) {
    int r = blockIdx.y;
    int e = blockIdx.x * 256 + threadIdx.x;
    const float* x = fin + (size_t)r * DMODEL;
    float acc = 0.f;
    for (int d = 0; d < DMODEL; ++d) acc += x[d] * tp[(size_t)d * DMODEL + e];
    feats[(size_t)r * DMODEL + e] = acc;
}

__global__ __launch_bounds__(256) void rownorm_kernel(float* __restrict__ feats) {
    int r = blockIdx.x, tid = threadIdx.x;
    float* x = feats + (size_t)r * DMODEL;
    float v0 = x[tid], v1 = x[tid + 256], v2 = x[tid + 512];
    __shared__ float red[256];
    red[tid] = v0 * v0 + v1 * v1 + v2 * v2;
    __syncthreads();
    for (int o = 128; o > 0; o >>= 1) {
        if (tid < o) red[tid] += red[tid + o];
        __syncthreads();
    }
    float inv = 1.0f / sqrtf(red[0]);
    x[tid] = v0 * inv; x[tid + 256] = v1 * inv; x[tid + 512] = v2 * inv;
}

// mean over n, renormalize, write d_out.  grid = 8 blocks.
__global__ __launch_bounds__(256) void meanout_kernel(const float* __restrict__ feats,
                                                      float* __restrict__ out) {
    int m = blockIdx.x, tid = threadIdx.x;
    float s[3];
#pragma unroll
    for (int jj = 0; jj < 3; ++jj) {
        int e = tid + jj * 256;
        float a = 0.f;
        for (int n = 0; n < 64; ++n) a += feats[((size_t)(m * 64 + n)) * DMODEL + e];
        s[jj] = a * (1.0f / 64.0f);
    }
    __shared__ float red[256];
    red[tid] = s[0] * s[0] + s[1] * s[1] + s[2] * s[2];
    __syncthreads();
    for (int o = 128; o > 0; o >>= 1) {
        if (tid < o) red[tid] += red[tid + o];
        __syncthreads();
    }
    float inv = 1.0f / sqrtf(red[0]);
#pragma unroll
    for (int jj = 0; jj < 3; ++jj) out[(size_t)m * DMODEL + tid + jj * 256] = s[jj] * inv;
}

// ---------------------------------------------------------------------------
// Host side
// ---------------------------------------------------------------------------
extern "C" void kernel_launch(void* const* d_in, const int* in_sizes, int n_in,
                              void* d_out, int out_size, void* d_ws, size_t ws_size,
                              hipStream_t stream) {
    const int*   text    = (const int*)  d_in[0];
    const float* visf    = (const float*)d_in[1];
    const float* tok_emb = (const float*)d_in[2];
    const float* pos_emb = (const float*)d_in[3];
    const float* ln1_w   = (const float*)d_in[4];
    const float* ln1_b   = (const float*)d_in[5];
    const float* in_w    = (const float*)d_in[6];
    const float* in_b    = (const float*)d_in[7];
    const float* out_w   = (const float*)d_in[8];
    const float* out_b   = (const float*)d_in[9];
    const float* ln2_w   = (const float*)d_in[10];
    const float* ln2_b   = (const float*)d_in[11];
    const float* fc_w    = (const float*)d_in[12];
    const float* fc_b    = (const float*)d_in[13];
    const float* pj_w    = (const float*)d_in[14];
    const float* pj_b    = (const float*)d_in[15];
    const float* lnf_w   = (const float*)d_in[16];
    const float* lnf_b   = (const float*)d_in[17];
    const float* tproj   = (const float*)d_in[18];

    // workspace bump allocator
    char* ws = (char*)d_ws;
    size_t off = 0;
    auto alloc = [&](size_t bytes) -> char* {
        char* p = ws + off;
        off = (off + bytes + 255) & ~(size_t)255;
        return p;
    };
    const long long n_inw  = 12LL * 2304 * 768;
    const long long n_outw = 12LL * 768 * 768;
    const long long n_fcw  = 12LL * 3072 * 768;
    const long long n_pjw  = 12LL * 768 * 3072;

    bf16_t* w_in  = (bf16_t*)alloc(n_inw  * 2);
    bf16_t* w_out = (bf16_t*)alloc(n_outw * 2);
    bf16_t* w_fc  = (bf16_t*)alloc(n_fcw  * 2);
    bf16_t* w_pj  = (bf16_t*)alloc(n_pjw  * 2);
    float*  h     = (float*) alloc((size_t)RROWS * DMODEL * 4);
    bf16_t* ybf   = (bf16_t*)alloc((size_t)RROWS * DMODEL * 2);
    bf16_t* qkvbf = (bf16_t*)alloc((size_t)RROWS * 2304 * 2);
    bf16_t* obf   = (bf16_t*)alloc((size_t)RROWS * DMODEL * 2);
    bf16_t* mlpbf = (bf16_t*)alloc((size_t)RROWS * 3072 * 2);
    int*    posy  = (int*)   alloc(64 * 4);
    int*    eot   = (int*)   alloc(64 * 4);
    float*  fin   = (float*) alloc((size_t)512 * DMODEL * 4);
    float*  feats = (float*) alloc((size_t)512 * DMODEL * 4);

    // weights -> bf16
    cvt_bf16_kernel<<<4096, 256, 0, stream>>>(in_w,  w_in,  n_inw);
    cvt_bf16_kernel<<<4096, 256, 0, stream>>>(out_w, w_out, n_outw);
    cvt_bf16_kernel<<<4096, 256, 0, stream>>>(fc_w,  w_fc,  n_fcw);
    cvt_bf16_kernel<<<4096, 256, 0, stream>>>(pj_w,  w_pj,  n_pjw);

    // prologue
    find_pos_kernel<<<1, 64, 0, stream>>>(text, posy, eot);
    embed_kernel<<<dim3(LSEQ, NSEQ), 256, 0, stream>>>(text, visf, tok_emb, pos_emb, posy, h);

    const int MB = RROWS / 64;   // 616 row-blocks
    for (int l = 0; l < 12; ++l) {
        ln_bf16_kernel<<<RROWS, 256, 0, stream>>>(h, ln1_w + l * 768, ln1_b + l * 768, ybf);
        gemm_wmma_kernel<0><<<dim3(2304 / 256, MB), 256, 0, stream>>>(
            ybf, w_in + (size_t)l * 2304 * 768, in_b + (size_t)l * 2304, qkvbf, 768, 2304);
        attention_kernel<<<dim3(NHEAD, NSEQ), 256, 0, stream>>>(qkvbf, obf);
        gemm_wmma_kernel<2><<<dim3(768 / 256, MB), 256, 0, stream>>>(
            obf, w_out + (size_t)l * 768 * 768, out_b + (size_t)l * 768, h, 768, 768);
        ln_bf16_kernel<<<RROWS, 256, 0, stream>>>(h, ln2_w + l * 768, ln2_b + l * 768, ybf);
        gemm_wmma_kernel<1><<<dim3(3072 / 256, MB), 256, 0, stream>>>(
            ybf, w_fc + (size_t)l * 3072 * 768, fc_b + (size_t)l * 3072, mlpbf, 768, 3072);
        gemm_wmma_kernel<2><<<dim3(768 / 256, MB), 256, 0, stream>>>(
            mlpbf, w_pj + (size_t)l * 768 * 3072, pj_b + (size_t)l * 768, h, 3072, 768);
    }

    // epilogue
    lnf_gather_kernel<<<512, 256, 0, stream>>>(h, eot, lnf_w, lnf_b, fin);
    proj_kernel<<<dim3(3, 512), 256, 0, stream>>>(fin, tproj, feats);
    rownorm_kernel<<<512, 256, 0, stream>>>(feats);
    meanout_kernel<<<8, 256, 0, stream>>>(feats, (float*)d_out);
}
